// MultiHeadAttention_41369124995637
// MI455X (gfx1250) — matrix operand order repaired and checked by reading, compiled, not run on hardware
//
#include <hip/hip_runtime.h>
#include <hip/hip_bf16.h>

typedef __bf16 bf16;
typedef __attribute__((ext_vector_type(16))) __bf16 v16bf;
typedef __attribute__((ext_vector_type(8)))  __bf16 v8bf;
typedef __attribute__((ext_vector_type(8)))  float  v8f;

#define B_  2
#define S_  2048
#define H_  1024
#define NH_ 16
#define HD_ 64

// ---- async global->LDS (gfx1250 GLOBAL_LOAD_ASYNC_TO_LDS_B128), guarded ----
#if __has_builtin(__builtin_amdgcn_global_load_async_to_lds_b128)
#define HAVE_ASYNC_LDS 1
// Builtin signature (from hipcc diagnostic): param0 = AS1 pointer to
// 'int __attribute__((vector_size(16)))', param1 = AS3 pointer to same,
// then i32 offset, i32 cpol.
typedef int v4i_ __attribute__((__vector_size__(16)));
typedef __attribute__((address_space(1))) v4i_ gv4i;
typedef __attribute__((address_space(3))) v4i_ lv4i;
__device__ __forceinline__ void async_cp16(const bf16* g, bf16* l) {
  __builtin_amdgcn_global_load_async_to_lds_b128(
      (gv4i*)g, (lv4i*)l, /*offset=*/0, /*cpol=*/0);
}
__device__ __forceinline__ void async_wait() {
#if __has_builtin(__builtin_amdgcn_s_wait_asynccnt)
  __builtin_amdgcn_s_wait_asynccnt(0);
#else
  asm volatile("s_wait_asynccnt 0x0" ::: "memory");
#endif
}
#else
#define HAVE_ASYNC_LDS 0
__device__ __forceinline__ void async_cp16(const bf16* g, bf16* l) {
  *(v8bf*)l = *(const v8bf*)g;   // fallback: VGPR round-trip
}
__device__ __forceinline__ void async_wait() {}
#endif

// D = A(16x32 bf16) * B(32x16 bf16) + C(16x16 f32)
__device__ __forceinline__ v8f wmma_bf16(v16bf a, v16bf b, v8f c) {
  return __builtin_amdgcn_wmma_f32_16x16x32_bf16(
      /*neg_a=*/false, a, /*neg_b=*/false, b,
      /*c_mod=*/(short)0, c, /*reuse_a=*/false, /*reuse_b=*/false);
}

__device__ __forceinline__ v8f zero_v8f() {
  v8f z = {0.f, 0.f, 0.f, 0.f, 0.f, 0.f, 0.f, 0.f};
  return z;
}

// A fragment: tile stored [m][k] row-major in LDS, `base` at (m0,k0) origin.
// lane<16 -> row=lane, k={0..7,16..23}; lane>=16 -> row=lane-16, k={8..15,24..31}
__device__ __forceinline__ v16bf frag_a(const bf16* base, int stride) {
  int lane = threadIdx.x & 31;
  int r    = lane & 15;
  int k0   = (lane >> 4) << 3;   // 0 or 8
  const bf16* p = base + r * stride + k0;
  v8bf lo = *(const v8bf*)(p);
  v8bf hi = *(const v8bf*)(p + 16);
  return __builtin_shufflevector(lo, hi, 0,1,2,3,4,5,6,7,8,9,10,11,12,13,14,15);
}

// B fragment: tile stored TRANSPOSED [n][k] in LDS, `base` at (n0,k0) origin.
// lane<16 -> n=lane, k={0..7,16..23}; lane>=16 -> n=lane-16, k={8..15,24..31}
__device__ __forceinline__ v16bf frag_b(const bf16* base, int stride) {
  int lane = threadIdx.x & 31;
  int n    = lane & 15;
  int k0   = (lane >> 4) << 3;   // 0 or 8
  const bf16* p = base + n * stride + k0;
  v8bf lo = *(const v8bf*)(p);
  v8bf hi = *(const v8bf*)(p + 16);
  return __builtin_shufflevector(lo, hi, 0,1,2,3,4,5,6,7,8,9,10,11,12,13,14,15);
}

// ---------------------------------------------------------------------------
// Kernel 1: X[4096,1024](f32) x W[1024,1024](f32) + b -> head-split bf16
// dst layout [B, NH, S, HD]. Block = 256 thr (8 waves), tile 128x128, K-tile 64.
// ---------------------------------------------------------------------------
__global__ __launch_bounds__(256)
void qkv_proj_kernel(const float* __restrict__ X, const float* __restrict__ W,
                     const float* __restrict__ bias, bf16* __restrict__ dst) {
  __shared__ __align__(16) bf16 As[128 * 72];   // [m][k], k-tile 64 (+8 pad)
  __shared__ __align__(16) bf16 Bs[128 * 72];   // [n][k] transposed
  const int t = threadIdx.x, lane = t & 31, w = t >> 5;
  const int wm = w & 3, wn = w >> 2;            // 4 waves on M, 2 on N
  const int m0 = blockIdx.x * 128, n0 = blockIdx.y * 128;

  v8f acc[2][4];
#pragma unroll
  for (int i = 0; i < 2; ++i)
#pragma unroll
    for (int j = 0; j < 4; ++j) acc[i][j] = zero_v8f();

  for (int kt = 0; kt < 16; ++kt) {
    __syncthreads();
    // stage A: 128x64 f32 -> bf16, row-major [m][k]
#pragma unroll
    for (int i = 0; i < 8; ++i) {
      int f4  = t + i * 256;                    // 2048 float4 chunks
      int row = f4 >> 4;
      int c4  = (f4 & 15) << 2;
      float4 x = *(const float4*)(X + (size_t)(m0 + row) * H_ + kt * 64 + c4);
      bf16* d = &As[row * 72 + c4];
      d[0] = (bf16)x.x; d[1] = (bf16)x.y; d[2] = (bf16)x.z; d[3] = (bf16)x.w;
    }
    // stage B: 64x128 f32 -> bf16, stored transposed [n][k]
#pragma unroll
    for (int i = 0; i < 8; ++i) {
      int f4 = t + i * 256;
      int kr = f4 >> 5;                         // 0..63
      int n4 = (f4 & 31) << 2;
      float4 x = *(const float4*)(W + (size_t)(kt * 64 + kr) * H_ + n0 + n4);
      Bs[(n4 + 0) * 72 + kr] = (bf16)x.x;
      Bs[(n4 + 1) * 72 + kr] = (bf16)x.y;
      Bs[(n4 + 2) * 72 + kr] = (bf16)x.z;
      Bs[(n4 + 3) * 72 + kr] = (bf16)x.w;
    }
    __syncthreads();
#pragma unroll
    for (int ks = 0; ks < 2; ++ks) {
      v16bf a0 = frag_a(&As[(wm * 32) * 72 + ks * 32], 72);
      v16bf a1 = frag_a(&As[(wm * 32 + 16) * 72 + ks * 32], 72);
#pragma unroll
      for (int nt = 0; nt < 4; ++nt) {
        v16bf b = frag_b(&Bs[(wn * 64 + nt * 16) * 72 + ks * 32], 72);
        acc[0][nt] = wmma_bf16(a0, b, acc[0][nt]);
        acc[1][nt] = wmma_bf16(a1, b, acc[1][nt]);
      }
    }
  }
  // epilogue: add bias, head-split scatter as bf16
  const int half = lane >> 4, col = lane & 15;
#pragma unroll
  for (int mt = 0; mt < 2; ++mt)
#pragma unroll
    for (int nt = 0; nt < 4; ++nt)
#pragma unroll
      for (int v = 0; v < 8; ++v) {
        int m = m0 + wm * 32 + mt * 16 + v + 8 * half;
        int n = n0 + wn * 64 + nt * 16 + col;
        float val = acc[mt][nt][v] + bias[n];
        int bb = m >> 11, ss = m & 2047, hh = n >> 6, dd = n & 63;
        dst[(((size_t)(bb * NH_ + hh)) * S_ + ss) * HD_ + dd] = (bf16)val;
      }
}

// ---------------------------------------------------------------------------
// Kernel 2: flash attention.  grid = (S/128, B*NH), block = 256 (8 waves).
// Q/K tiles staged with async global->LDS when available.
// ---------------------------------------------------------------------------
__global__ __launch_bounds__(256)
void flash_attn_kernel(const bf16* __restrict__ Q, const bf16* __restrict__ K,
                       const bf16* __restrict__ V, const float* __restrict__ mask,
                       bf16* __restrict__ ctx) {
  __shared__ __align__(16) bf16 Qs[128 * 72];      // [q][d]
  __shared__ __align__(16) bf16 Ks[64 * 72];       // [sk][d]  (Bt for QK^T)
  __shared__ __align__(16) bf16 Vts[64 * 72];      // [d][sk]  (Bt for PV)
  __shared__ __align__(16) bf16 Ps[8 * 16 * 72];   // per-wave P [16][64]

  const int t = threadIdx.x, lane = t & 31, w = t >> 5;
  const int q0 = blockIdx.x * 128;
  const int bh = blockIdx.y;
  const int bb = bh >> 4, hh = bh & 15;
  const size_t headbase = (size_t)bh * S_ * HD_;
  const int half = lane >> 4, col = lane & 15;

  // stage Q tile 128x64 (async -> waited before first use inside the loop)
#pragma unroll
  for (int i = 0; i < 4; ++i) {
    int c = t + i * 256;                 // 1024 16B chunks
    int row = c >> 3;
    int c8 = (c & 7) << 3;
    async_cp16(Q + headbase + (size_t)(q0 + row) * HD_ + c8, &Qs[row * 72 + c8]);
  }

  float m_st[8], l_st[8];
  v8f o[4];
#pragma unroll
  for (int v = 0; v < 8; ++v) { m_st[v] = -1e30f; l_st[v] = 0.f; }
#pragma unroll
  for (int d = 0; d < 4; ++d) o[d] = zero_v8f();

  bf16* Pw = &Ps[w * 16 * 72];

  for (int kt = 0; kt < 32; ++kt) {
    __syncthreads();
    // stage K chunk [64][64] via async copy
#pragma unroll
    for (int i = 0; i < 2; ++i) {
      int c = t + i * 256;
      int row = c >> 3;
      int c8 = (c & 7) << 3;
      async_cp16(K + headbase + (size_t)(kt * 64 + row) * HD_ + c8,
                 &Ks[row * 72 + c8]);
    }
    // stage V chunk transposed -> Vt[d][sk] (needs shuffle: VGPR path)
#pragma unroll
    for (int i = 0; i < 2; ++i) {
      int c = t + i * 256;
      int sk = c >> 3;
      int d8 = (c & 7) << 3;
      v8bf vv = *(const v8bf*)(V + headbase + (size_t)(kt * 64 + sk) * HD_ + d8);
#pragma unroll
      for (int j = 0; j < 8; ++j) Vts[(d8 + j) * 72 + sk] = vv[j];
    }
    async_wait();          // ASYNCcnt==0: Q (first iter) + K landed in LDS
    __syncthreads();

    if (kt < 31)  // speculative prefetch of next K chunk -> global_prefetch_b8
      __builtin_prefetch(K + headbase + (size_t)((kt + 1) * 64 + (t >> 2)) * HD_ +
                             (t & 3) * 16, 0, 0);

    // scores S = Q(16x64) * K^T(64x64): A from Qs, B from Ks ([n=sk][k=d])
    v8f sacc[4];
#pragma unroll
    for (int nt = 0; nt < 4; ++nt) sacc[nt] = zero_v8f();
#pragma unroll
    for (int ks = 0; ks < 2; ++ks) {
      v16bf a = frag_a(&Qs[(w * 16) * 72 + ks * 32], 72);
#pragma unroll
      for (int nt = 0; nt < 4; ++nt) {
        v16bf b = frag_b(&Ks[(nt * 16) * 72 + ks * 32], 72);
        sacc[nt] = wmma_bf16(a, b, sacc[nt]);
      }
    }
    // scale 1/sqrt(64) + additive mask (broadcast over q rows)
#pragma unroll
    for (int nt = 0; nt < 4; ++nt) {
      float mv = mask[(size_t)bb * S_ + kt * 64 + nt * 16 + col];
#pragma unroll
      for (int v = 0; v < 8; ++v) sacc[nt][v] = sacc[nt][v] * 0.125f + mv;
    }
    // online softmax: a row lives across 16 lanes (same half), fixed v
#pragma unroll
    for (int v = 0; v < 8; ++v) {
      float r = fmaxf(fmaxf(sacc[0][v], sacc[1][v]), fmaxf(sacc[2][v], sacc[3][v]));
      r = fmaxf(r, __shfl_xor(r, 1, 32));
      r = fmaxf(r, __shfl_xor(r, 2, 32));
      r = fmaxf(r, __shfl_xor(r, 4, 32));
      r = fmaxf(r, __shfl_xor(r, 8, 32));
      float mn = fmaxf(m_st[v], r);
      float corr = __expf(m_st[v] - mn);
      m_st[v] = mn;
      l_st[v] *= corr;
#pragma unroll
      for (int dt = 0; dt < 4; ++dt) o[dt][v] *= corr;
      float rs = 0.f;
#pragma unroll
      for (int nt = 0; nt < 4; ++nt) {
        float p = __expf(sacc[nt][v] - mn);
        sacc[nt][v] = p;
        rs += p;
      }
      rs += __shfl_xor(rs, 1, 32);
      rs += __shfl_xor(rs, 2, 32);
      rs += __shfl_xor(rs, 4, 32);
      rs += __shfl_xor(rs, 8, 32);
      l_st[v] += rs;
    }
    // P -> per-wave LDS [16][64] so it can be re-read in A-fragment layout
#pragma unroll
    for (int nt = 0; nt < 4; ++nt)
#pragma unroll
      for (int v = 0; v < 8; ++v)
        Pw[(v + 8 * half) * 72 + nt * 16 + col] = (bf16)sacc[nt][v];

    // O += P(16x64) * V(64x64): A from Pw, B from Vts ([n=d][k=sk])
#pragma unroll
    for (int k4 = 0; k4 < 2; ++k4) {
      v16bf a = frag_a(&Pw[k4 * 32], 72);
#pragma unroll
      for (int dt = 0; dt < 4; ++dt) {
        v16bf b = frag_b(&Vts[(dt * 16) * 72 + k4 * 32], 72);
        o[dt] = wmma_bf16(a, b, o[dt]);
      }
    }
  }
  // normalize and write ctx as bf16 in [B, S, H] (heads re-merged)
#pragma unroll
  for (int v = 0; v < 8; ++v) {
    float inv = (l_st[v] > 0.f) ? (1.f / l_st[v]) : 0.f;
    int srow = q0 + w * 16 + v + 8 * half;
#pragma unroll
    for (int dt = 0; dt < 4; ++dt)
      ctx[((size_t)bb * S_ + srow) * H_ + hh * HD_ + dt * 16 + col] =
          (bf16)(o[dt][v] * inv);
  }
}

// ---------------------------------------------------------------------------
// Kernel 3: ctx[4096,1024](bf16) x Wo[1024,1024](f32->bf16) + bo -> f32 out
// A tile staged with async global->LDS when available.
// ---------------------------------------------------------------------------
__global__ __launch_bounds__(256)
void out_proj_kernel(const bf16* __restrict__ Cx, const float* __restrict__ W,
                     const float* __restrict__ bias, float* __restrict__ out) {
  __shared__ __align__(16) bf16 As[128 * 72];
  __shared__ __align__(16) bf16 Bs[128 * 72];
  const int t = threadIdx.x, lane = t & 31, w = t >> 5;
  const int wm = w & 3, wn = w >> 2;
  const int m0 = blockIdx.x * 128, n0 = blockIdx.y * 128;

  v8f acc[2][4];
#pragma unroll
  for (int i = 0; i < 2; ++i)
#pragma unroll
    for (int j = 0; j < 4; ++j) acc[i][j] = zero_v8f();

  for (int kt = 0; kt < 16; ++kt) {
    __syncthreads();
    // stage A (already bf16): 1024 16B chunks via async copy
#pragma unroll
    for (int i = 0; i < 4; ++i) {
      int c = t + i * 256;
      int row = c >> 3;
      int c8 = (c & 7) << 3;
      async_cp16(Cx + (size_t)(m0 + row) * H_ + kt * 64 + c8,
                 &As[row * 72 + c8]);
    }
    // stage B transposed with f32->bf16 convert
#pragma unroll
    for (int i = 0; i < 8; ++i) {
      int f4 = t + i * 256;
      int kr = f4 >> 5;
      int n4 = (f4 & 31) << 2;
      float4 x = *(const float4*)(W + (size_t)(kt * 64 + kr) * H_ + n0 + n4);
      Bs[(n4 + 0) * 72 + kr] = (bf16)x.x;
      Bs[(n4 + 1) * 72 + kr] = (bf16)x.y;
      Bs[(n4 + 2) * 72 + kr] = (bf16)x.z;
      Bs[(n4 + 3) * 72 + kr] = (bf16)x.w;
    }
    async_wait();
    __syncthreads();
#pragma unroll
    for (int ks = 0; ks < 2; ++ks) {
      v16bf a0 = frag_a(&As[(wm * 32) * 72 + ks * 32], 72);
      v16bf a1 = frag_a(&As[(wm * 32 + 16) * 72 + ks * 32], 72);
#pragma unroll
      for (int nt = 0; nt < 4; ++nt) {
        v16bf b = frag_b(&Bs[(wn * 64 + nt * 16) * 72 + ks * 32], 72);
        acc[0][nt] = wmma_bf16(a0, b, acc[0][nt]);
        acc[1][nt] = wmma_bf16(a1, b, acc[1][nt]);
      }
    }
  }
  const int half = lane >> 4, col = lane & 15;
#pragma unroll
  for (int mt = 0; mt < 2; ++mt)
#pragma unroll
    for (int nt = 0; nt < 4; ++nt)
#pragma unroll
      for (int v = 0; v < 8; ++v) {
        int m = m0 + wm * 32 + mt * 16 + v + 8 * half;
        int n = n0 + wn * 64 + nt * 16 + col;
        out[(size_t)m * H_ + n] = acc[mt][nt][v] + bias[n];
      }
}

// ---------------------------------------------------------------------------
extern "C" void kernel_launch(void* const* d_in, const int* in_sizes, int n_in,
                              void* d_out, int out_size, void* d_ws, size_t ws_size,
                              hipStream_t stream) {
  (void)in_sizes; (void)n_in; (void)out_size; (void)ws_size;
  const float* hs   = (const float*)d_in[0];
  const float* mask = (const float*)d_in[1];
  const float* Wq   = (const float*)d_in[2];
  const float* bq   = (const float*)d_in[3];
  const float* Wk   = (const float*)d_in[4];
  const float* bk   = (const float*)d_in[5];
  const float* Wv   = (const float*)d_in[6];
  const float* bv   = (const float*)d_in[7];
  const float* Wo   = (const float*)d_in[8];
  const float* bo   = (const float*)d_in[9];
  float* out = (float*)d_out;

  // workspace: Q,K,V head-split bf16 + ctx bf16  (4 x 8 MB = 32 MB)
  const size_t per = (size_t)B_ * NH_ * S_ * HD_;
  bf16* Qws = (bf16*)d_ws;
  bf16* Kws = Qws + per;
  bf16* Vws = Kws + per;
  bf16* Cws = Vws + per;

  dim3 blk(256);
  dim3 gproj(32, 8);      // 4096/128 x 1024/128
  qkv_proj_kernel<<<gproj, blk, 0, stream>>>(hs, Wq, bq, Qws);
  qkv_proj_kernel<<<gproj, blk, 0, stream>>>(hs, Wk, bk, Kws);
  qkv_proj_kernel<<<gproj, blk, 0, stream>>>(hs, Wv, bv, Vws);

  dim3 gattn(S_ / 128, B_ * NH_);   // 16 x 32
  flash_attn_kernel<<<gattn, blk, 0, stream>>>(Qws, Kws, Vws, mask, Cws);

  out_proj_kernel<<<gproj, blk, 0, stream>>>(Cws, Wo, bo, out);
}
